// RelativeBiasTransformer_40922448396393
// MI455X (gfx1250) — compile-verified, matrix operand order checked by reading, not run on hardware
//
#include <hip/hip_runtime.h>
#include <hip/hip_bf16.h>

#define TL 4
#define BB 8
#define MM 1024
#define HH 768
#define NHH 12
#define HDD 64
#define MTOT (BB * MM)

typedef __bf16 bf16_t;
typedef bf16_t v8bf __attribute__((ext_vector_type(8)));
typedef bf16_t v16bf __attribute__((ext_vector_type(16)));
typedef float v8f __attribute__((ext_vector_type(8)));
typedef int v4i __attribute__((ext_vector_type(4)));

#if defined(__gfx1250__) && __has_builtin(__builtin_amdgcn_global_load_async_to_lds_b128)
#define HAVE_ASYNC_LDS 1
typedef __attribute__((address_space(1))) v4i* g_v4i_ptr;
typedef __attribute__((address_space(3))) v4i* l_v4i_ptr;
#endif

__device__ __forceinline__ bf16_t f2bf(float f) {
  unsigned u = __builtin_bit_cast(unsigned, f);
  u += 0x7FFFu + ((u >> 16) & 1u);
  unsigned short h = (unsigned short)(u >> 16);
  return __builtin_bit_cast(bf16_t, h);
}
__device__ __forceinline__ float bf2f(bf16_t b) {
  unsigned short h = __builtin_bit_cast(unsigned short, b);
  unsigned u = ((unsigned)h) << 16;
  return __builtin_bit_cast(float, u);
}
__device__ __forceinline__ v16bf cat8(v8bf lo, v8bf hi) {
  return __builtin_shufflevector(lo, hi, 0,1,2,3,4,5,6,7,8,9,10,11,12,13,14,15);
}
__device__ __forceinline__ v8f wmma_bf16(v16bf a, v16bf b, v8f c) {
  // D = A(16x32 bf16) x B(32x16 bf16) + C(16x16 f32)
  return __builtin_amdgcn_wmma_f32_16x16x32_bf16(false, a, false, b, (short)0, c, false, false);
}

// 16-byte global -> LDS copy: async path on CDNA5, sync fallback.
__device__ __forceinline__ void copy16_g2l(const bf16_t* g, bf16_t* l) {
#if defined(HAVE_ASYNC_LDS)
  __builtin_amdgcn_global_load_async_to_lds_b128((g_v4i_ptr)(g), (l_v4i_ptr)(l), 0, 0);
#else
  *(v8bf*)l = *(const v8bf*)g;
#endif
}
__device__ __forceinline__ void wait_async_lds() {
#if defined(HAVE_ASYNC_LDS)
#if __has_builtin(__builtin_amdgcn_s_wait_asynccnt)
  __builtin_amdgcn_s_wait_asynccnt(0);
#else
  asm volatile("s_wait_asynccnt 0" ::: "memory");
#endif
#endif
}

// ---------------------------------------------------------------------------
// Relative bias precompute (layer- and head-invariant):
// bias[b][q][key] = ov_emb[min(overlap,10)] - alpha * clip(lp[b][key],-30,0)
// ---------------------------------------------------------------------------
__global__ void bias_kernel(const int* __restrict__ overlap, const float* __restrict__ lp,
                            const float* __restrict__ ov_emb, const float* __restrict__ alpha_p,
                            float* __restrict__ bias) {
  const float alpha = alpha_p[0];
  const size_t total = (size_t)BB * MM * MM;
  for (size_t i = (size_t)blockIdx.x * blockDim.x + threadIdx.x; i < total;
       i += (size_t)gridDim.x * blockDim.x) {
    int key = (int)(i & (MM - 1));
    int b = (int)(i >> 20);  // M*M == 2^20
    int ov = overlap[i];
    ov = ov < 10 ? ov : 10;
    float lpc = fminf(fmaxf(lp[(size_t)b * MM + key], -30.f), 0.f);
    bias[i] = ov_emb[ov] - alpha * lpc;
  }
}

// ---------------------------------------------------------------------------
// fp32 [K][N] -> bf16 transposed [N][K] weight conversion.
// Grid: (N/256, K); coalesced reads along n, one-time cost per layer.
// ---------------------------------------------------------------------------
__global__ void cvt_t_kernel(const float* __restrict__ src, bf16_t* __restrict__ dst,
                             int K, int N) {
  const int n = blockIdx.x * 256 + threadIdx.x;
  const int k = blockIdx.y;
  dst[(size_t)n * K + k] = f2bf(src[(size_t)k * N + n]);
}

// ---------------------------------------------------------------------------
// LayerNorm over H=768, one row per block (256 threads, 3 elems each), bf16 out
// ---------------------------------------------------------------------------
__global__ void ln_kernel(const float* __restrict__ x, const float* __restrict__ sc,
                          const float* __restrict__ bs, bf16_t* __restrict__ out) {
  const int t = threadIdx.x;
  const size_t row = blockIdx.x;
  const float* xr = x + row * HH;
  float v[3];
  float s = 0.f;
#pragma unroll
  for (int i = 0; i < 3; i++) { v[i] = xr[t + i * 256]; s += v[i]; }
  __shared__ float red[256];
  red[t] = s;
  __syncthreads();
  for (int o = 128; o > 0; o >>= 1) { if (t < o) red[t] += red[t + o]; __syncthreads(); }
  const float mu = red[0] * (1.f / HH);
  __syncthreads();
  float q = 0.f;
#pragma unroll
  for (int i = 0; i < 3; i++) { float d = v[i] - mu; q += d * d; }
  red[t] = q;
  __syncthreads();
  for (int o = 128; o > 0; o >>= 1) { if (t < o) red[t] += red[t + o]; __syncthreads(); }
  const float inv = rsqrtf(red[0] * (1.f / HH) + 1e-5f);
#pragma unroll
  for (int i = 0; i < 3; i++) {
    int col = t + i * 256;
    out[row * HH + col] = f2bf((v[i] - mu) * inv * sc[col] + bs[col]);
  }
}

// ---------------------------------------------------------------------------
// Tiled bf16 WMMA GEMM: out = act(A[MxK] @ Wt[NxK]^T + bias) (+ resid)
// Both tiles contiguous [row][k] -> pure async b128 staging, no transposes.
// Block tile 128x128, 8 waves each own 32(M)x64(N) = 2x4 WMMA accumulators.
// ---------------------------------------------------------------------------
__device__ __forceinline__ void stage_tiles(const bf16_t* __restrict__ A,
                                            const bf16_t* __restrict__ Wt,
                                            bf16_t* ldsA, bf16_t* ldsB,
                                            int bm, int bn, int k0, int K, int t) {
#pragma unroll
  for (int i = 0; i < 2; i++) {
    int lv = t + i * 256;
    int row = lv >> 2, kc = (lv & 3) * 8;
    copy16_g2l(A + (size_t)(bm + row) * K + k0 + kc, &ldsA[row * 32 + kc]);
  }
#pragma unroll
  for (int i = 0; i < 2; i++) {
    int lv = t + i * 256;
    int row = lv >> 2, kc = (lv & 3) * 8;
    copy16_g2l(Wt + (size_t)(bn + row) * K + k0 + kc, &ldsB[row * 32 + kc]);
  }
}

template <bool HASB, bool HASR, bool GELU, bool OUTF, bool QKV>
__global__ void gemm_bf16_kernel(const bf16_t* __restrict__ A, const bf16_t* __restrict__ Wt,
                                 const float* __restrict__ bias, const float* __restrict__ resid,
                                 float* __restrict__ outF, bf16_t* __restrict__ outB,
                                 bf16_t* __restrict__ vt, int N, int K) {
  __shared__ bf16_t ldsA[2][128 * 32];  // [m][k]
  __shared__ bf16_t ldsB[2][128 * 32];  // [n][k]
  const int t = threadIdx.x, lane = t & 31, wave = t >> 5;
  const int hl = lane & 15, lh = lane >> 4;
  const int koff = lh * 8;
  const int bm = blockIdx.y * 128, bn = blockIdx.x * 128;
  const int m0w = (wave >> 1) * 32, n0w = (wave & 1) * 64;

  v8f zero = {};
  v8f acc[2][4];
#pragma unroll
  for (int i = 0; i < 2; i++)
#pragma unroll
    for (int j = 0; j < 4; j++) acc[i][j] = zero;

  // prologue: stage first K-slab into buffer 0
  stage_tiles(A, Wt, ldsA[0], ldsB[0], bm, bn, 0, K, t);

  for (int k0 = 0; k0 < K; k0 += 32) {
    const int cur = (k0 >> 5) & 1;
    wait_async_lds();        // this wave's async copies into buf[cur] are done
    __syncthreads();         // everyone's copies visible; prior readers of buf[cur^1] done

    if (k0 + 32 < K)         // overlap next slab's fetch with this slab's WMMAs
      stage_tiles(A, Wt, ldsA[cur ^ 1], ldsB[cur ^ 1], bm, bn, k0 + 32, K, t);

    v16bf af[2], bfr[4];
#pragma unroll
    for (int i = 0; i < 2; i++) {
      const bf16_t* p = &ldsA[cur][(m0w + i * 16 + hl) * 32 + koff];
      af[i] = cat8(*(const v8bf*)p, *(const v8bf*)(p + 16));
    }
#pragma unroll
    for (int j = 0; j < 4; j++) {
      const bf16_t* p = &ldsB[cur][(n0w + j * 16 + hl) * 32 + koff];
      bfr[j] = cat8(*(const v8bf*)p, *(const v8bf*)(p + 16));
    }
#pragma unroll
    for (int i = 0; i < 2; i++)
#pragma unroll
      for (int j = 0; j < 4; j++) acc[i][j] = wmma_bf16(af[i], bfr[j], acc[i][j]);
  }

  // epilogue: bias / gelu / residual / store (compile-time specialized)
#pragma unroll
  for (int i = 0; i < 2; i++)
#pragma unroll
    for (int j = 0; j < 4; j++)
#pragma unroll
      for (int r = 0; r < 8; r++) {
        int row = bm + m0w + i * 16 + r + lh * 8;
        int col = bn + n0w + j * 16 + hl;
        float v = acc[i][j][r];
        if (HASB) v += bias[col];
        if (GELU) v = 0.5f * v * (1.f + erff(v * 0.70710678f));
        if (HASR) v += resid[(size_t)row * N + col];
        if (OUTF) {
          outF[(size_t)row * N + col] = v;
        } else if (QKV && col >= 2 * HH) {
          // V head-columns stored transposed: vt[b][h][d][m]
          int hc = col - 2 * HH;
          int hd = hc >> 6, d = hc & 63;
          int b_ = row >> 10, m_ = row & (MM - 1);
          vt[(((size_t)b_ * NHH + hd) * HDD + d) * MM + m_] = f2bf(v);
        } else {
          outB[(size_t)row * N + col] = f2bf(v);
        }
      }
}

// ---------------------------------------------------------------------------
// Flash attention with precomputed relative bias. Barrier-free waves.
// Grid: (M/128, NH, B). Block: 256 (8 waves). Q-tile 128 (16 rows/wave),
// key chunks of 64 with online softmax. Q/K from qkv bf16 [B][M][3H];
// V from pre-transposed vt bf16 [B][NH][HD][M]; bias f32 [B][M][M].
// ---------------------------------------------------------------------------
__global__ void attn_kernel(const bf16_t* __restrict__ qkv, const bf16_t* __restrict__ vt,
                            const float* __restrict__ bias, bf16_t* __restrict__ o) {
  const int H3 = 3 * HH;
  const int qt = blockIdx.x, hh = blockIdx.y, bi = blockIdx.z;
  const int t = threadIdx.x, lane = t & 31, wave = t >> 5;
  const int hl = lane & 15, lh = lane >> 4;
  const int koff = lh * 8;
  const float scale = 0.125f; // 1/sqrt(64)

  __shared__ float Sb[8][16 * 64];      // per-wave scores
  __shared__ bf16_t Pb[8][16 * 64];     // per-wave probabilities (bf16)
  __shared__ float mSt[128], lSt[128], rfSt[128];

  // wave-local state init (no barrier needed: same-wave LDS ordering)
  if (lane < 16) { mSt[wave * 16 + lane] = -1e30f; lSt[wave * 16 + lane] = 0.f; }

  const int q0 = qt * 128;
  const int qrow = q0 + wave * 16 + hl;
  const bf16_t* qrp = qkv + ((size_t)bi * MM + qrow) * H3 + hh * 64;
  v16bf qa[2];
  qa[0] = cat8(*(const v8bf*)(qrp + koff), *(const v8bf*)(qrp + 16 + koff));
  qa[1] = cat8(*(const v8bf*)(qrp + 32 + koff), *(const v8bf*)(qrp + 48 + koff));

  const bf16_t* vbase = vt + ((size_t)bi * NHH + hh) * HDD * MM;
  const float* bbase = bias + ((size_t)bi * MM + q0 + wave * 16) * MM;

  v8f oacc[4];
  {
    v8f zero = {};
#pragma unroll
    for (int d = 0; d < 4; d++) oacc[d] = zero;
  }

  for (int kc0 = 0; kc0 < MM; kc0 += 64) {
    // S = Q @ K^T over 4 key subtiles, + scale + precomputed bias -> Sb
#pragma unroll
    for (int nt = 0; nt < 4; nt++) {
      const int keyn = kc0 + nt * 16 + hl;
      const bf16_t* krp = qkv + ((size_t)bi * MM + keyn) * H3 + HH + hh * 64;
      v16bf kb0 = cat8(*(const v8bf*)(krp + koff), *(const v8bf*)(krp + 16 + koff));
      v16bf kb1 = cat8(*(const v8bf*)(krp + 32 + koff), *(const v8bf*)(krp + 48 + koff));
      v8f s = {};
      s = wmma_bf16(qa[0], kb0, s);
      s = wmma_bf16(qa[1], kb1, s);
#pragma unroll
      for (int r = 0; r < 8; r++) {
        int rl = r + lh * 8;
        int key = kc0 + nt * 16 + hl;
        Sb[wave][rl * 64 + nt * 16 + hl] = s[r] * scale + bbase[(size_t)rl * MM + key];
      }
    }

    // online softmax (wave-local): lane pair (hl, lh) owns row hl, cols lh*32..+31
    {
      const int row = hl;
      const int cbase = row * 64 + lh * 32;
      float mloc = -1e30f;
#pragma unroll
      for (int j = 0; j < 32; j++) mloc = fmaxf(mloc, Sb[wave][cbase + j]);
      mloc = fmaxf(mloc, __shfl_xor(mloc, 16, 32));
      const float mprev = mSt[wave * 16 + row];
      const float mnew = fmaxf(mprev, mloc);
      const float rf = __expf(mprev - mnew);
      float lsum = 0.f;
#pragma unroll
      for (int j = 0; j < 32; j++) {
        float e = __expf(Sb[wave][cbase + j] - mnew);
        Pb[wave][cbase + j] = f2bf(e);
        lsum += e;
      }
      lsum += __shfl_xor(lsum, 16, 32);
      if (lh == 0) {
        mSt[wave * 16 + row] = mnew;
        lSt[wave * 16 + row] = lSt[wave * 16 + row] * rf + lsum;
        rfSt[wave * 16 + row] = rf;
      }
    }

    // rescale O accumulators by exp(m_prev - m_new)
#pragma unroll
    for (int dt = 0; dt < 4; dt++)
#pragma unroll
      for (int r = 0; r < 8; r++) oacc[dt][r] *= rfSt[wave * 16 + r + lh * 8];

    // O += P @ V  (V fragments straight from transposed global layout)
    {
      const bf16_t* prp = &Pb[wave][hl * 64];
      v16bf pa0 = cat8(*(const v8bf*)(prp + koff), *(const v8bf*)(prp + 16 + koff));
      v16bf pa1 = cat8(*(const v8bf*)(prp + 32 + koff), *(const v8bf*)(prp + 48 + koff));
#pragma unroll
      for (int dt = 0; dt < 4; dt++) {
        const bf16_t* vrp = vbase + (size_t)(dt * 16 + hl) * MM + kc0;
        v16bf vb0 = cat8(*(const v8bf*)(vrp + koff), *(const v8bf*)(vrp + 16 + koff));
        v16bf vb1 = cat8(*(const v8bf*)(vrp + 32 + koff), *(const v8bf*)(vrp + 48 + koff));
        oacc[dt] = wmma_bf16(pa0, vb0, oacc[dt]);
        oacc[dt] = wmma_bf16(pa1, vb1, oacc[dt]);
      }
    }
  }

  // normalize and store o (bf16)
#pragma unroll
  for (int dt = 0; dt < 4; dt++)
#pragma unroll
    for (int r = 0; r < 8; r++) {
      int rl = r + lh * 8;
      int qq = q0 + wave * 16 + rl;
      float linv = 1.f / lSt[wave * 16 + rl];
      o[((size_t)bi * MM + qq) * HH + hh * 64 + dt * 16 + hl] = f2bf(oacc[dt][r] * linv);
    }
}

// ---------------------------------------------------------------------------
// HT head: out[b,:] = sum_m softmax(-clip(lp))_m * xln[b,m,:]
// ---------------------------------------------------------------------------
__global__ void ht_kernel(const bf16_t* __restrict__ xln, const float* __restrict__ lp,
                          float* __restrict__ out) {
  const int bi = blockIdx.x, t = threadIdx.x;
  __shared__ float w[MM];
  __shared__ float red[256];
  const float* lpb = lp + (size_t)bi * MM;
  float loc = -1e30f;
#pragma unroll
  for (int i = 0; i < 4; i++) {
    float lpc = fminf(fmaxf(lpb[t + i * 256], -30.f), 0.f);
    float val = -lpc;
    w[t + i * 256] = val;
    loc = fmaxf(loc, val);
  }
  red[t] = loc;
  __syncthreads();
  for (int o = 128; o > 0; o >>= 1) { if (t < o) red[t] = fmaxf(red[t], red[t + o]); __syncthreads(); }
  const float mx = red[0];
  __syncthreads();
  float ls = 0.f;
#pragma unroll
  for (int i = 0; i < 4; i++) {
    float e = __expf(w[t + i * 256] - mx);
    w[t + i * 256] = e;
    ls += e;
  }
  red[t] = ls;
  __syncthreads();
  for (int o = 128; o > 0; o >>= 1) { if (t < o) red[t] += red[t + o]; __syncthreads(); }
  const float inv = 1.f / red[0];
  __syncthreads();
#pragma unroll
  for (int j = 0; j < 3; j++) {
    int col = t + j * 256;
    float acc = 0.f;
    for (int m = 0; m < MM; m++)
      acc += w[m] * bf2f(xln[((size_t)bi * MM + m) * HH + col]);
    out[(size_t)bi * HH + col] = acc * inv;
  }
}

// ---------------------------------------------------------------------------
// host orchestration
// ---------------------------------------------------------------------------
extern "C" void kernel_launch(void* const* d_in, const int* in_sizes, int n_in,
                              void* d_out, int out_size, void* d_ws, size_t ws_size,
                              hipStream_t stream) {
  (void)in_sizes; (void)n_in; (void)out_size; (void)ws_size;
  const float* x      = (const float*)d_in[0];
  const int*   ovl    = (const int*)d_in[1];
  const float* lp     = (const float*)d_in[2];
  const float* ln1_s  = (const float*)d_in[3];
  const float* ln1_b  = (const float*)d_in[4];
  const float* Wqkv   = (const float*)d_in[5];
  const float* Wout   = (const float*)d_in[6];
  const float* bout   = (const float*)d_in[7];
  const float* ln2_s  = (const float*)d_in[8];
  const float* ln2_b  = (const float*)d_in[9];
  const float* W1     = (const float*)d_in[10];
  const float* b1     = (const float*)d_in[11];
  const float* W2     = (const float*)d_in[12];
  const float* b2     = (const float*)d_in[13];
  const float* lno_s  = (const float*)d_in[14];
  const float* lno_b  = (const float*)d_in[15];
  const float* ov_emb = (const float*)d_in[16];
  const float* ht_a   = (const float*)d_in[17];

  char* ws = (char*)d_ws;
  size_t off = 0;
  auto alloc = [&](size_t bytes) -> void* {
    void* p = ws + off;
    off += (bytes + 255) & ~(size_t)255;
    return p;
  };
  const int H = HH;
  bf16_t* wqkv_t = (bf16_t*)alloc((size_t)3 * H * H * 2);  // [3H][H]
  bf16_t* wout_t = (bf16_t*)alloc((size_t)H * H * 2);      // [H][H]
  bf16_t* w1_t   = (bf16_t*)alloc((size_t)4 * H * H * 2);  // [4H][H]
  bf16_t* w2_t   = (bf16_t*)alloc((size_t)H * 4 * H * 2);  // [H][4H]
  bf16_t* r_b    = (bf16_t*)alloc((size_t)MTOT * H * 2);
  bf16_t* qkv_b  = (bf16_t*)alloc((size_t)MTOT * 3 * H * 2);
  bf16_t* vt_b   = (bf16_t*)alloc((size_t)MTOT * H * 2);   // [B][NH][HD][M]
  bf16_t* o_b    = (bf16_t*)alloc((size_t)MTOT * H * 2);
  bf16_t* g_b    = (bf16_t*)alloc((size_t)MTOT * 4 * H * 2);
  float*  x_work = (float*)alloc((size_t)MTOT * H * 4);
  float*  bias_f = (float*)alloc((size_t)BB * MM * MM * 4); // 33.5 MB

  const dim3 blk(256);
  // bias is layer- and head-invariant: compute once
  bias_kernel<<<8192, blk, 0, stream>>>(ovl, lp, ov_emb, ht_a, bias_f);

  const float* xin = x;
  for (int t = 0; t < TL; t++) {
    // weight convert + transpose: [K][N] f32 -> [N][K] bf16
    cvt_t_kernel<<<dim3(3 * H / 256, H), blk, 0, stream>>>(Wqkv + (size_t)t * H * 3 * H, wqkv_t, H, 3 * H);
    cvt_t_kernel<<<dim3(H / 256, H), blk, 0, stream>>>(Wout + (size_t)t * H * H, wout_t, H, H);
    cvt_t_kernel<<<dim3(4 * H / 256, H), blk, 0, stream>>>(W1 + (size_t)t * H * 4 * H, w1_t, H, 4 * H);
    cvt_t_kernel<<<dim3(H / 256, 4 * H), blk, 0, stream>>>(W2 + (size_t)t * 4 * H * H, w2_t, 4 * H, H);

    ln_kernel<<<MTOT, blk, 0, stream>>>(xin, ln1_s + t * H, ln1_b + t * H, r_b);
    // qkv = r @ Wqkv -> q,k into qkv_b (bf16), v transposed into vt_b
    gemm_bf16_kernel<false, false, false, false, true>
        <<<dim3(3 * H / 128, MTOT / 128), blk, 0, stream>>>(
            r_b, wqkv_t, nullptr, nullptr, nullptr, qkv_b, vt_b, 3 * H, H);
    attn_kernel<<<dim3(MM / 128, NHH, BB), blk, 0, stream>>>(qkv_b, vt_b, bias_f, o_b);
    // x = x + o @ Wout + bout             -> f32
    gemm_bf16_kernel<true, true, false, true, false>
        <<<dim3(H / 128, MTOT / 128), blk, 0, stream>>>(
            o_b, wout_t, bout + t * H, xin, x_work, nullptr, nullptr, H, H);
    ln_kernel<<<MTOT, blk, 0, stream>>>(x_work, ln2_s + t * H, ln2_b + t * H, r_b);
    // g = gelu(h @ W1 + b1)               -> bf16
    gemm_bf16_kernel<true, false, true, false, false>
        <<<dim3(4 * H / 128, MTOT / 128), blk, 0, stream>>>(
            r_b, w1_t, b1 + t * 4 * H, nullptr, nullptr, g_b, nullptr, 4 * H, H);
    // x = x + g @ W2 + b2                 -> f32
    gemm_bf16_kernel<true, true, false, true, false>
        <<<dim3(H / 128, MTOT / 128), blk, 0, stream>>>(
            g_b, w2_t, b2 + t * H, x_work, x_work, nullptr, nullptr, H, 4 * H);
    xin = x_work;
  }
  ln_kernel<<<MTOT, blk, 0, stream>>>(x_work, lno_s, lno_b, r_b);
  ht_kernel<<<BB, blk, 0, stream>>>(r_b, lp, (float*)d_out);
}
